// VectorQuantizer2_2568390443495
// MI455X (gfx1250) — compile-verified, matrix operand order checked by reading, not run on hardware
//
#include <hip/hip_runtime.h>
#include <hip/hip_bf16.h>

typedef __bf16 bf16_t;
typedef __attribute__((ext_vector_type(8)))  __bf16 v8bf;
typedef __attribute__((ext_vector_type(16))) __bf16 v16bf;
typedef __attribute__((ext_vector_type(8)))  float  v8f;

#define BQ   16
#define CD   256
#define HH   64
#define WW   64
#define KC   2048
#define NROW (BQ*HH*WW)      // 65536
#define ROWS 16
#define DSTRIDE 2052         // 2048 + 4 pad -> conflict-free LDS banks
#define ZSTRIDE 264          // 256 + 8 bf16 pad (16B aligned rows)
#define VQBETA 0.25f

static __device__ inline v16bf make16(v8bf a, v8bf b) {
  return __builtin_shufflevector(a, b, 0,1,2,3,4,5,6,7,8,9,10,11,12,13,14,15);
}

// ---------------------------------------------------------------------------
// Kernel 1: codebook -> bf16 hi/lo split + ||e_k||^2
// ---------------------------------------------------------------------------
__global__ __launch_bounds__(256) void vq_prep_codebook(
    const float* __restrict__ cb, bf16_t* __restrict__ cbhi,
    bf16_t* __restrict__ cblo, float* __restrict__ enorm) {
  const int k = blockIdx.x;
  const int c = threadIdx.x;
  float v = cb[(size_t)k * CD + c];
  bf16_t hi = (bf16_t)v;
  bf16_t lo = (bf16_t)(v - (float)hi);
  cbhi[(size_t)k * CD + c] = hi;
  cblo[(size_t)k * CD + c] = lo;
  __shared__ float red[256];
  red[c] = v * v;
  __syncthreads();
  for (int s = 128; s > 0; s >>= 1) {
    if (c < s) red[c] += red[c + s];
    __syncthreads();
  }
  if (c == 0) enorm[k] = red[0];
}

// ---------------------------------------------------------------------------
// Kernel 2: fused distances (WMMA bf16x3) + argmin + softmax(-|d|) + z_q
// One block = 16 rows (same b,h; w0..w0+15). Full K=2048 panel kept in LDS.
// ---------------------------------------------------------------------------
__global__ __launch_bounds__(256, 1) void vq_main(
    const float* __restrict__ feats, const float* __restrict__ cb,
    const bf16_t* __restrict__ cbhi, const bf16_t* __restrict__ cblo,
    const float* __restrict__ enorm,
    float* __restrict__ zq_out, float* __restrict__ idx_out,
    float* __restrict__ probs_out, float* __restrict__ rowmin_out) {

  __shared__ __align__(16) bf16_t zhi[ROWS * ZSTRIDE];
  __shared__ __align__(16) bf16_t zlo[ROWS * ZSTRIDE];
  __shared__ float partial[ROWS][16];
  __shared__ float znorm[ROWS];
  __shared__ int   shidx[ROWS];
  __shared__ float dbuf[ROWS][DSTRIDE];          // ~131 KB, CDNA5 WGP LDS

  const int tid = threadIdx.x;
  const int n0  = blockIdx.x * ROWS;
  const int b   = n0 >> 12;                      // / (H*W)
  const int rem = n0 & 4095;
  const int h   = rem >> 6;
  const int w0  = rem & 63;                      // multiple of 16

  // ---- stage z rows (fp32 -> bf16 hi/lo) + row norms (deterministic) ----
  {
    const int w = tid & 15;
    const int cb0 = tid >> 4;
    float acc = 0.f;
    for (int c = cb0; c < CD; c += 16) {
      float v = feats[(((size_t)b * CD + c) * HH + h) * WW + w0 + w];
      bf16_t hb = (bf16_t)v;
      zhi[w * ZSTRIDE + c] = hb;
      zlo[w * ZSTRIDE + c] = (bf16_t)(v - (float)hb);
      acc += v * v;
    }
    partial[w][cb0] = acc;
  }
  __syncthreads();
  if (tid < ROWS) {
    float s = 0.f;
    #pragma unroll
    for (int i = 0; i < 16; ++i) s += partial[tid][i];
    znorm[tid] = s;
  }
  __syncthreads();

  // ---- GEMM phase: each wave owns 16 of the 128 column tiles ----
  const int wv    = tid >> 5;
  const int lane  = tid & 31;
  const int mrow  = lane & 15;                   // A row / B col / D col
  const int kbase = (lane >> 4) * 8;             // K sub-offset per ISA layout
  const int mb    = (lane >> 4) * 8;             // D row base

  for (int jt = wv; jt < KC / 16; jt += 8) {
    const int col0 = jt * 16;
    const int bcol = col0 + mrow;
    const bf16_t* __restrict__ bh = cbhi + (size_t)bcol * CD;
    const bf16_t* __restrict__ bl = cblo + (size_t)bcol * CD;
    v8f acc = {};
    for (int kc = 0; kc < CD; kc += 32) {
      const bf16_t* za = &zhi[mrow * ZSTRIDE + kc + kbase];
      const bf16_t* zb = &zlo[mrow * ZSTRIDE + kc + kbase];
      v16bf ahi = make16(*(const v8bf*)za, *(const v8bf*)(za + 16));
      v16bf alo = make16(*(const v8bf*)zb, *(const v8bf*)(zb + 16));
      v16bf bhi = make16(*(const v8bf*)(bh + kc + kbase),
                         *(const v8bf*)(bh + kc + 16 + kbase));
      v16bf blo = make16(*(const v8bf*)(bl + kc + kbase),
                         *(const v8bf*)(bl + kc + 16 + kbase));
      acc = __builtin_amdgcn_wmma_f32_16x16x32_bf16(false, ahi, false, bhi,
                                                    (short)0, acc, false, false);
      acc = __builtin_amdgcn_wmma_f32_16x16x32_bf16(false, ahi, false, blo,
                                                    (short)0, acc, false, false);
      acc = __builtin_amdgcn_wmma_f32_16x16x32_bf16(false, alo, false, bhi,
                                                    (short)0, acc, false, false);
    }
    const float en = enorm[bcol];
    #pragma unroll
    for (int v = 0; v < 8; ++v) {
      const int m = v + mb;                      // D: M = vgpr + 8*(lane>=16)
      dbuf[m][bcol] = znorm[m] + en - 2.0f * acc[v];
    }
  }
  __syncthreads();

  // ---- per-row reductions: argmin(d), min|d|, sum exp; normalize in LDS ----
  #pragma unroll
  for (int rr = 0; rr < 2; ++rr) {
    const int r = wv * 2 + rr;
    float bmin = 3.4e38f, babs = 3.4e38f;
    int   bidx = 0;
    for (int k = lane; k < KC; k += 32) {
      float d = dbuf[r][k];
      float ad = fabsf(d);
      if (d < bmin || (d == bmin && k < bidx)) { bmin = d; bidx = k; }
      babs = fminf(babs, ad);
    }
    #pragma unroll
    for (int off = 16; off > 0; off >>= 1) {
      float om = __shfl_xor(bmin, off, 32);
      int   oi = __shfl_xor(bidx, off, 32);
      float oa = __shfl_xor(babs, off, 32);
      if (om < bmin || (om == bmin && oi < bidx)) { bmin = om; bidx = oi; }
      babs = fminf(babs, oa);
    }
    float s = 0.f;
    for (int k = lane; k < KC; k += 32)
      s += __expf(babs - fabsf(dbuf[r][k]));
    #pragma unroll
    for (int off = 16; off > 0; off >>= 1) s += __shfl_xor(s, off, 32);
    const float inv = 1.0f / s;
    for (int k = lane; k < KC; k += 32)
      dbuf[r][k] = __expf(babs - fabsf(dbuf[r][k])) * inv;
    if (lane == 0) {
      shidx[r] = bidx;
      idx_out[n0 + r]    = (float)bidx;          // d_out is all-float
      rowmin_out[n0 + r] = bmin;                 // = ||z - e_q||^2
    }
  }
  __syncthreads();

  // ---- coalesced write-out: probs [B,K,H,W] with w in lanes ----
  {
    const int w  = tid & 15;
    const int kb = tid >> 4;
    const size_t base = (((size_t)b * KC) * HH + h) * WW + w0 + w;
    for (int k = kb; k < KC; k += 16)
      probs_out[base + (size_t)k * (HH * WW)] = dbuf[w][k];
  }
  // ---- z_q gather (straight-through forward value = codebook row) ----
  {
    const int w  = tid & 15;
    const int c0 = tid >> 4;
    const float* __restrict__ crow = cb + (size_t)shidx[w] * CD;
    const size_t base = (((size_t)b * CD) * HH + h) * WW + w0 + w;
    for (int c = c0; c < CD; c += 16)
      zq_out[base + (size_t)c * (HH * WW)] = crow[c];
  }
}

// ---------------------------------------------------------------------------
// Kernel 3: loss = (1+beta) * sum(rowmin) / (N*C)   (deterministic reduce)
// ---------------------------------------------------------------------------
__global__ __launch_bounds__(256) void vq_loss(
    const float* __restrict__ rowmin, float* __restrict__ loss) {
  __shared__ float red[256];
  float s = 0.f;
  for (int i = threadIdx.x; i < NROW; i += 256) s += rowmin[i];
  red[threadIdx.x] = s;
  __syncthreads();
  for (int st = 128; st > 0; st >>= 1) {
    if (threadIdx.x < st) red[threadIdx.x] += red[threadIdx.x + st];
    __syncthreads();
  }
  if (threadIdx.x == 0)
    loss[0] = (1.0f + VQBETA) * red[0] / (float)((size_t)NROW * CD);
}

// ---------------------------------------------------------------------------
extern "C" void kernel_launch(void* const* d_in, const int* in_sizes, int n_in,
                              void* d_out, int out_size, void* d_ws, size_t ws_size,
                              hipStream_t stream) {
  const float* feats = (const float*)d_in[0];   // [16,256,64,64]
  const float* cb    = (const float*)d_in[1];   // [2048,256]

  float* out   = (float*)d_out;
  float* zq    = out;                                   // 16*256*64*64
  float* idx   = zq + (size_t)BQ * CD * HH * WW;        // 65536
  float* loss  = idx + NROW;                            // 1
  float* probs = loss + 1;                              // 16*2048*64*64

  char* ws = (char*)d_ws;
  bf16_t* cbhi  = (bf16_t*)ws;                                   // 1 MB
  bf16_t* cblo  = (bf16_t*)(ws + (size_t)KC * CD * 2);           // 1 MB
  float* enorm  = (float*)(ws + (size_t)KC * CD * 4);            // 8 KB
  float* rowmin = (float*)(ws + (size_t)KC * CD * 4 + KC * 4);   // 256 KB

  vq_prep_codebook<<<KC, 256, 0, stream>>>(cb, cbhi, cblo, enorm);
  vq_main<<<NROW / ROWS, 256, 0, stream>>>(feats, cb, cbhi, cblo, enorm,
                                           zq, idx, probs, rowmin);
  vq_loss<<<1, 256, 0, stream>>>(rowmin, loss);
}
